// GPT2Attention_40527311405240
// MI455X (gfx1250) — matrix operand order repaired and checked by reading, not compiled, not run on hardware
//
#include <hip/hip_runtime.h>

// ---------------------------------------------------------------------------
// GPT-2 attention block for MI455X (gfx1250, wave32, WMMA f32_16x16x32_f16).
//   1) cast x -> f16, transpose+cast w_attn -> [3D][D] f16, w_proj -> [D][D] f16
//   2) QKV GEMM (64x64/wave WMMA), bias, scatter Q[B,H,S,HD], K[B,H,S,HD], Vt[B,H,HD,S]
//   3) flash attention: per-wave 16-row Q tile, 32 keys/iter, online softmax,
//      P restaged via LDS into WMMA A-layout
//   4) projection GEMM (64x64/wave WMMA) + bias -> fp32 output
// GEMM inner loop: 2 base pointers/lane + 24-bit load immediates; GEMM kernels
// use __launch_bounds__(256,1) to unlock the full VGPR budget (no C spills).
// ---------------------------------------------------------------------------

typedef _Float16 f16;
typedef __attribute__((ext_vector_type(16))) _Float16 v16h;
typedef __attribute__((ext_vector_type(8)))  _Float16 v8h;
typedef __attribute__((ext_vector_type(8)))  float    v8f;

#define B_  2
#define S_  2048
#define D_  1024
#define H_  16
#define HD_ 64

union V16U { v16h v; v8h h[2]; };

static __device__ __forceinline__ v8f wmma_f16(v16h a, v16h b, v8f c) {
  // 8 args: (neg_a, A, neg_b, B, c_mod, C, reuse_a, reuse_b)
  return __builtin_amdgcn_wmma_f32_16x16x32_f16(false, a, false, b, (short)0, c,
                                                false, false);
}

// ---------------------------------------------------------------- pre-passes
__global__ void k_cast_f16(const float* __restrict__ in, f16* __restrict__ out, int n) {
  int i = blockIdx.x * blockDim.x + threadIdx.x;
  if (i < n) out[i] = (f16)in[i];
}

// w [K,N] fp32 row-major -> wT [N,K] f16 (B-matrix fragments become contiguous)
__global__ void k_transpose_f16(const float* __restrict__ w, f16* __restrict__ wT,
                                int K, int N) {
  int i = blockIdx.x * blockDim.x + threadIdx.x;
  if (i < K * N) {
    int k = i / N, n = i % N;
    wT[(size_t)n * K + k] = (f16)w[i];
  }
}

// ------------------------------------------------------------- QKV GEMM
// C[4096,3072] = xh[4096,1024] @ w_attn ; per-wave 64x64 tile (4x4 WMMA).
__global__ void __launch_bounds__(256, 1)
k_gemm_qkv(const f16* __restrict__ A, const f16* __restrict__ BT,
           const float* __restrict__ bias,
           f16* __restrict__ Qb, f16* __restrict__ Kb, f16* __restrict__ Vt) {
  const int lane = threadIdx.x & 31;
  const int kg   = lane >> 4;      // which 16-lane half
  const int ln   = lane & 15;
  const int wid  = blockIdx.x * (blockDim.x >> 5) + (threadIdx.x >> 5);
  const int NT   = (3 * D_) / 64;  // 48 col tiles
  const int m0   = (wid / NT) * 64;
  const int n0   = (wid % NT) * 64;

  // One base pointer per operand; mt/nt strides (32KB) fold into the 24-bit
  // immediate offset of global_load_b128.
  const f16* pa = A  + (size_t)(m0 + ln) * D_ + 16 * kg;
  const f16* pb = BT + (size_t)(n0 + ln) * D_ + 16 * kg;

  v8f acc[4][4] = {};
  for (int kk = 0; kk < D_; kk += 32) {
    V16U a[4], b[4];
#pragma unroll
    for (int mt = 0; mt < 4; ++mt) {
      a[mt].h[0] = *(const v8h*)(pa + mt * (16 * D_));       // K = kk+16*kg..
      a[mt].h[1] = *(const v8h*)(pa + mt * (16 * D_) + 8);   // K = kk+16*kg+8..
    }
#pragma unroll
    for (int nt = 0; nt < 4; ++nt) {
      b[nt].v = *(const v16h*)(pb + nt * (16 * D_));         // K = kk+16*kg..
      if (kk + 128 < D_)
        __builtin_prefetch(pb + nt * (16 * D_) + 128, 0, 3); // global_prefetch_b8
    }
#pragma unroll
    for (int mt = 0; mt < 4; ++mt)
#pragma unroll
      for (int nt = 0; nt < 4; ++nt)
        acc[mt][nt] = wmma_f16(a[mt].v, b[nt].v, acc[mt][nt]);
    pa += 32;
    pb += 32;
  }

#pragma unroll
  for (int nt = 0; nt < 4; ++nt) {
    const int col  = n0 + nt * 16 + ln;   // 0..3071
    const float bv = bias[col];
    const int which = col >> 10;          // 0=Q 1=K 2=V
    const int d  = col & (D_ - 1);
    const int hh = d >> 6;
    const int hd = d & 63;
#pragma unroll
    for (int mt = 0; mt < 4; ++mt) {
#pragma unroll
      for (int v = 0; v < 8; ++v) {
        const int row = m0 + mt * 16 + v + 8 * kg;   // C layout: M = v + 8*kg
        const int bb  = row >> 11;                   // / 2048
        const int ss  = row & (S_ - 1);
        const f16 val = (f16)(acc[mt][nt][v] + bv);
        if (which == 0)
          Qb[((size_t)(bb * H_ + hh) * S_ + ss) * HD_ + hd] = val;
        else if (which == 1)
          Kb[((size_t)(bb * H_ + hh) * S_ + ss) * HD_ + hd] = val;
        else
          Vt[((size_t)(bb * H_ + hh) * HD_ + hd) * S_ + ss] = val;
      }
    }
  }
}

// ------------------------------------------------------------- flash attention
// One wave per (b, h, 16-row q tile); 32 keys per iteration.
__global__ void k_attn(const f16* __restrict__ Qb, const f16* __restrict__ Kb,
                       const f16* __restrict__ Vt, f16* __restrict__ Yh) {
  __shared__ f16 pstage[8][16 * 32];   // per-wave P staging (C-layout -> A-layout)
  const int lane = threadIdx.x & 31;
  const int kg   = lane >> 4;
  const int ln   = lane & 15;
  const int w    = threadIdx.x >> 5;
  const int wid  = blockIdx.x * (blockDim.x >> 5) + w;
  const int qt = wid & (S_ / 16 - 1);
  const int h  = (wid >> 7) & (H_ - 1);
  const int b  = wid >> 11;
  const int q0 = qt * 16;

  const f16* Qp = Qb + (size_t)(b * H_ + h) * S_ * HD_;
  const f16* Vp = Vt + (size_t)(b * H_ + h) * HD_ * S_;

  // Q A-fragments (row = q0 + ln), two K=32 chunks of HD=64
  V16U aq[2];
#pragma unroll
  for (int c = 0; c < 2; ++c) {
    const f16* p = Qp + (size_t)(q0 + ln) * HD_ + c * 32;
    aq[c].h[0] = *(const v8h*)(p + 16 * kg);
    aq[c].h[1] = *(const v8h*)(p + 16 * kg + 8);
  }

  float m[8], l[8];
  v8f acc[4] = {};                      // 16x64 output accumulator (4 C-tiles)
#pragma unroll
  for (int v = 0; v < 8; ++v) { m[v] = -INFINITY; l[v] = 0.f; }

  // advancing base pointers (stride 32 keys/iter)
  const f16* pk = Kb + (size_t)(b * H_ + h) * S_ * HD_ + (size_t)ln * HD_ + 16 * kg;
  const f16* pv = Vp + (size_t)ln * S_ + 16 * kg;

  const float scale = 0.125f;           // 1/sqrt(64)
  const int nkb = (q0 + 16 + 31) >> 5;  // causal: key <= q0+15
  for (int kb = 0; kb < nkb; ++kb) {
    const int k0 = kb * 32;

    // S = Q @ K^T for 2 key sub-tiles of 16  (K rows stride 16*HD folds to imm)
    v8f sa[2];
#pragma unroll
    for (int t = 0; t < 2; ++t) {
      V16U bk[2];
#pragma unroll
      for (int c = 0; c < 2; ++c)
        bk[c].v = *(const v16h*)(pk + t * (16 * HD_) + c * 32);
      v8f s = {};
      s = wmma_f16(aq[0].v, bk[0].v, s);
      s = wmma_f16(aq[1].v, bk[1].v, s);
      sa[t] = s;
    }

    const bool needMask = (k0 + 31 > q0);  // uniform per wave
    float p0[8], p1[8];
#pragma unroll
    for (int v = 0; v < 8; ++v) {
      const int qrow = q0 + v + 8 * kg;
      float s0 = sa[0][v] * scale;
      float s1 = sa[1][v] * scale;
      if (needMask) {
        if (k0 + ln > qrow)      s0 = -INFINITY;
        if (k0 + 16 + ln > qrow) s1 = -INFINITY;
      }
      // row max across the 16 lanes of this half
      float mx = fmaxf(s0, s1);
      mx = fmaxf(mx, __shfl_xor(mx, 1));
      mx = fmaxf(mx, __shfl_xor(mx, 2));
      mx = fmaxf(mx, __shfl_xor(mx, 4));
      mx = fmaxf(mx, __shfl_xor(mx, 8));
      const float mnew = fmaxf(m[v], mx);
      const float a0 = __expf(s0 - mnew);
      const float a1 = __expf(s1 - mnew);
      float sum = a0 + a1;
      sum += __shfl_xor(sum, 1);
      sum += __shfl_xor(sum, 2);
      sum += __shfl_xor(sum, 4);
      sum += __shfl_xor(sum, 8);
      const float alpha = __expf(m[v] - mnew);
      l[v] = l[v] * alpha + sum;
      m[v] = mnew;
#pragma unroll
      for (int n = 0; n < 4; ++n) acc[n][v] *= alpha;
      p0[v] = a0; p1[v] = a1;
    }

    // C-layout P -> LDS -> A-layout fragment (LDS ops in-order within a wave)
#pragma unroll
    for (int v = 0; v < 8; ++v) {
      pstage[w][(v + 8 * kg) * 32 + ln]      = (f16)p0[v];
      pstage[w][(v + 8 * kg) * 32 + 16 + ln] = (f16)p1[v];
    }
    __builtin_amdgcn_wave_barrier();
    V16U ap;
    {
      const f16* p = &pstage[w][ln * 32];
      ap.h[0] = *(const v8h*)(p + 16 * kg);
      ap.h[1] = *(const v8h*)(p + 16 * kg + 8);
    }
    __builtin_amdgcn_wave_barrier();

    // O += P @ V ; Vt[hd][s] makes B fragments contiguous along keys
#pragma unroll
    for (int n = 0; n < 4; ++n) {
      V16U bv;
      bv.v = *(const v16h*)(pv + n * (16 * S_));  // 64KB stride folds to imm
      acc[n] = wmma_f16(ap.v, bv.v, acc[n]);
    }

    pk += 32 * HD_;   // next 32 keys (rows of K)
    pv += 32;         // next 32 keys (cols of Vt)
  }

  // epilogue: y[b, qrow, h*64 + col] (f16 for projection GEMM)
#pragma unroll
  for (int v = 0; v < 8; ++v) {
    const int qrow = q0 + v + 8 * kg;
    const float inv = 1.0f / l[v];
    f16* yrow = Yh + ((size_t)(b * S_) + qrow) * D_ + h * HD_;
#pragma unroll
    for (int n = 0; n < 4; ++n)
      yrow[n * 16 + ln] = (f16)(acc[n][v] * inv);
  }
}

// ------------------------------------------------------------- projection GEMM
// out[4096,1024] fp32 = Yh @ w_proj + bias ; per-wave 64x64 tile (4x4 WMMA).
__global__ void __launch_bounds__(256, 1)
k_gemm_proj(const f16* __restrict__ A, const f16* __restrict__ BT,
            const float* __restrict__ bias, float* __restrict__ out) {
  const int lane = threadIdx.x & 31;
  const int kg   = lane >> 4;
  const int ln   = lane & 15;
  const int wid  = blockIdx.x * (blockDim.x >> 5) + (threadIdx.x >> 5);
  const int NT   = D_ / 64;    // 16 col tiles
  const int m0   = (wid / NT) * 64;
  const int n0   = (wid % NT) * 64;

  const f16* pa = A  + (size_t)(m0 + ln) * D_ + 16 * kg;
  const f16* pb = BT + (size_t)(n0 + ln) * D_ + 16 * kg;

  v8f acc[4][4] = {};
  for (int kk = 0; kk < D_; kk += 32) {
    V16U a[4], b[4];
#pragma unroll
    for (int mt = 0; mt < 4; ++mt) {
      a[mt].h[0] = *(const v8h*)(pa + mt * (16 * D_));
      a[mt].h[1] = *(const v8h*)(pa + mt * (16 * D_) + 8);
    }
#pragma unroll
    for (int nt = 0; nt < 4; ++nt) {
      b[nt].v = *(const v16h*)(pb + nt * (16 * D_));
      if (kk + 128 < D_)
        __builtin_prefetch(pb + nt * (16 * D_) + 128, 0, 3);
    }
#pragma unroll
    for (int mt = 0; mt < 4; ++mt)
#pragma unroll
      for (int nt = 0; nt < 4; ++nt)
        acc[mt][nt] = wmma_f16(a[mt].v, b[nt].v, acc[mt][nt]);
    pa += 32;
    pb += 32;
  }

#pragma unroll
  for (int nt = 0; nt < 4; ++nt) {
    const int col = n0 + nt * 16 + ln;
    const float bv = bias[col];
#pragma unroll
    for (int mt = 0; mt < 4; ++mt)
#pragma unroll
      for (int v = 0; v < 8; ++v) {
        const int row = m0 + mt * 16 + v + 8 * kg;
        out[(size_t)row * D_ + col] = acc[mt][nt][v] + bv;
      }
  }
}

// --------------------------------------------------------------------- launch
extern "C" void kernel_launch(void* const* d_in, const int* in_sizes, int n_in,
                              void* d_out, int out_size, void* d_ws, size_t ws_size,
                              hipStream_t stream) {
  const float* x      = (const float*)d_in[0];   // [B,S,D]
  const float* w_attn = (const float*)d_in[1];   // [D,3D]
  const float* b_attn = (const float*)d_in[2];   // [3D]
  const float* w_proj = (const float*)d_in[3];   // [D,D]
  const float* b_proj = (const float*)d_in[4];   // [D]
  float* out = (float*)d_out;                    // [B,S,D] fp32

  char* ws = (char*)d_ws;
  const size_t NX   = (size_t)B_ * S_ * D_;      // 4,194,304
  const size_t NWA  = (size_t)D_ * 3 * D_;       // 3,145,728
  const size_t NWP  = (size_t)D_ * D_;           // 1,048,576
  f16* xh  = (f16*)(ws);                               // 8 MB
  f16* waT = (f16*)(ws + NX * 2);                      // 6 MB
  f16* wpT = (f16*)(ws + NX * 2 + NWA * 2);            // 2 MB
  f16* Qb  = (f16*)(ws + NX * 2 + NWA * 2 + NWP * 2);  // 8 MB
  f16* Kb  = Qb + NX;                                  // 8 MB
  f16* Vt  = Kb + NX;                                  // 8 MB
  f16* Yh  = Vt + NX;                                  // 8 MB  (total 48 MB)

  // 1) precision / layout pre-pass
  k_cast_f16<<<(int)(NX / 256), 256, 0, stream>>>(x, xh, (int)NX);
  k_transpose_f16<<<(int)(NWA / 256), 256, 0, stream>>>(w_attn, waT, D_, 3 * D_);
  k_transpose_f16<<<(int)(NWP / 256), 256, 0, stream>>>(w_proj, wpT, D_, D_);

  // 2) QKV GEMM: (4096/64)*(3072/64) = 3072 waves, 8 waves/block
  k_gemm_qkv<<<3072 / 8, 256, 0, stream>>>(xh, waT, b_attn, Qb, Kb, Vt);

  // 3) flash attention: B*H*(S/16) = 4096 waves
  k_attn<<<4096 / 8, 256, 0, stream>>>(Qb, Kb, Vt, Yh);

  // 4) projection: (4096/64)*(1024/64) = 1024 waves
  k_gemm_proj<<<1024 / 8, 256, 0, stream>>>(Yh, wpT, b_proj, out);

  (void)in_sizes; (void)n_in; (void)out_size; (void)ws_size;
}
